// GNNEncoder_29910152249704
// MI455X (gfx1250) — matrix operand order repaired and checked by reading, not compiled, not run on hardware
//
#include <hip/hip_runtime.h>

typedef __attribute__((ext_vector_type(16))) __bf16 v16bf;
typedef __attribute__((ext_vector_type(8)))  float  v8f;

#define NN   50000
#define NE   800000
#define DIM  128
#define EDIM 32

// ---- order-preserving float<->int encoding for atomic max ----
__device__ __forceinline__ int enc_f(float f) {
    int i = __float_as_int(f);
    return i >= 0 ? i : (i ^ 0x7fffffff);
}
__device__ __forceinline__ float dec_f(int i) {
    return __int_as_float(i >= 0 ? i : (i ^ 0x7fffffff));
}

// ---- f32 -> bf16 cast ----
__global__ void k_cast_bf16(const float* __restrict__ in, __bf16* __restrict__ out, int n) {
    int i = blockIdx.x * blockDim.x + threadIdx.x;
    if (i < n) out[i] = (__bf16)in[i];
}

// ---- transpose + cast: in[rows x cols] f32 -> out[cols x rows] bf16 ----
__global__ void k_transpose_cast(const float* __restrict__ in, __bf16* __restrict__ out,
                                 int rows, int cols) {
    int i = blockIdx.x * blockDim.x + threadIdx.x;
    if (i < rows * cols) {
        int r = i / cols, c = i - r * cols;
        out[c * rows + r] = (__bf16)in[i];
    }
}

// ---- per-layer init: hbuf=0, smax=-inf(encoded), den=0 ----
__global__ void k_init(int* __restrict__ smaxi, float* __restrict__ den,
                       float* __restrict__ hbuf, int nh) {
    int i = blockIdx.x * blockDim.x + threadIdx.x;
    if (i < NN * DIM) hbuf[i] = 0.f;
    if (i < nh) { smaxi[i] = (int)0x80000000; den[i] = 0.f; }
}

// ---- node GEMM: out[nrows x 128] = A_bf16[nrows x 128] @ Bt^T + bias ----
// Bt is pre-transposed [128 out][128 k]. One wave computes 16 rows x 128 cols.
__global__ void __launch_bounds__(256) k_gemm_node(
    const __bf16* __restrict__ A, const __bf16* __restrict__ Bt,
    const float* __restrict__ bias, float* __restrict__ out, int nrows) {
    int wave = threadIdx.x >> 5, lane = threadIdx.x & 31;
    int r0 = (blockIdx.x * 8 + wave) * 16;
    if (r0 >= nrows) return;
    int mrow = lane & 15, hi = lane >> 4;
    const __bf16* arow = A + (size_t)(r0 + mrow) * DIM;
    v16bf a[4];
#pragma unroll
    for (int ks = 0; ks < 4; ++ks) {
        const __bf16* p = arow + ks * 32 + hi * 8;
#pragma unroll
        for (int i = 0; i < 8; ++i) { a[ks][i] = p[i]; a[ks][8 + i] = p[16 + i]; }
    }
#pragma unroll
    for (int nt = 0; nt < 8; ++nt) {
        int col = nt * 16 + mrow;
        float bv = bias[col];
        v8f acc = {bv, bv, bv, bv, bv, bv, bv, bv};
#pragma unroll
        for (int ks = 0; ks < 4; ++ks) {
            const __bf16* bp = Bt + (size_t)col * DIM + ks * 32 + hi * 8;
            v16bf b;
#pragma unroll
            for (int i = 0; i < 8; ++i) { b[i] = bp[i]; b[8 + i] = bp[16 + i]; }
            acc = __builtin_amdgcn_wmma_f32_16x16x32_bf16(false, a[ks], false, b,
                                                          (short)0, acc, false, false);
        }
#pragma unroll
        for (int r = 0; r < 8; ++r)
            out[(size_t)(r0 + r + hi * 8) * DIM + col] = acc[r];
    }
}

// ---- fused edge kernel: WMMA edge projection (K=32) -> LDS, then
//      m = xl[src]+xr[dst]+e, LeakyReLU, per-head att dot -> sbuf[E x H] ----
__global__ void __launch_bounds__(128) k_edge_score(
    const __bf16* __restrict__ eabf, const __bf16* __restrict__ Wet,
    const float* __restrict__ xl, const float* __restrict__ xr,
    const int* __restrict__ src, const int* __restrict__ dst,
    const float* __restrict__ att, float* __restrict__ sbuf, int H, int shift) {
    __shared__ float tile[4][16][DIM];
    int wave = threadIdx.x >> 5, lane = threadIdx.x & 31;
    int e0 = (blockIdx.x * 4 + wave) * 16;
    int mrow = lane & 15, hi = lane >> 4;

    const __bf16* ep = eabf + (size_t)(e0 + mrow) * EDIM + hi * 8;
    v16bf a;
#pragma unroll
    for (int i = 0; i < 8; ++i) { a[i] = ep[i]; a[8 + i] = ep[16 + i]; }

#pragma unroll
    for (int nt = 0; nt < 8; ++nt) {
        const __bf16* bp = Wet + (size_t)(nt * 16 + mrow) * EDIM + hi * 8;
        v16bf b;
#pragma unroll
        for (int i = 0; i < 8; ++i) { b[i] = bp[i]; b[8 + i] = bp[16 + i]; }
        v8f acc = {};
        acc = __builtin_amdgcn_wmma_f32_16x16x32_bf16(false, a, false, b,
                                                      (short)0, acc, false, false);
#pragma unroll
        for (int r = 0; r < 8; ++r)
            tile[wave][r + hi * 8][nt * 16 + mrow] = acc[r];
    }
    __syncthreads();

    int le = lane >> 1, half = lane & 1;
    int eg = e0 + le;
    const float* xlp = xl + (size_t)src[eg] * DIM + half * 64;
    const float* xrp = xr + (size_t)dst[eg] * DIM + half * 64;
    const float* ap  = att + half * 64;
    float sacc[4] = {0.f, 0.f, 0.f, 0.f};
#pragma unroll 8
    for (int i = 0; i < 64; ++i) {
        int ch = half * 64 + i;
        float m = xlp[i] + xrp[i] + tile[wave][le][ch];
        m = m > 0.f ? m : 0.2f * m;
        sacc[ch >> shift] += m * ap[i];
    }
    for (int h = 0; h < H; ++h) sacc[h] += __shfl_xor(sacc[h], 1);
    if (half == 0)
        for (int h = 0; h < H; ++h) sbuf[(size_t)eg * H + h] = sacc[h];
}

// ---- segment max over dst (encoded-int atomic max) ----
__global__ void k_seg_max(const float* __restrict__ sbuf, const int* __restrict__ dst,
                          int* __restrict__ smaxi, int H) {
    int i = blockIdx.x * blockDim.x + threadIdx.x;
    if (i >= NE * H) return;
    int e = i / H, h = i - e * H;
    atomicMax(&smaxi[dst[e] * H + h], enc_f(sbuf[i]));
}

// ---- softmax denominator ----
__global__ void k_seg_den(const float* __restrict__ sbuf, const int* __restrict__ dst,
                          const int* __restrict__ smaxi, float* __restrict__ den, int H) {
    int i = blockIdx.x * blockDim.x + threadIdx.x;
    if (i >= NE * H) return;
    int e = i / H, h = i - e * H;
    int idx = dst[e] * H + h;
    atomicAdd(&den[idx], __expf(sbuf[i] - dec_f(smaxi[idx])));
}

// ---- weighted aggregation: hbuf[dst] += xl[src] * alpha  (32 lanes/edge, 4 ch/lane) ----
__global__ void __launch_bounds__(256) k_aggregate(
    const float* __restrict__ sbuf, const int* __restrict__ src, const int* __restrict__ dst,
    const int* __restrict__ smaxi, const float* __restrict__ den,
    const float* __restrict__ xl, float* __restrict__ hbuf, int H, int shift) {
    int i = blockIdx.x * blockDim.x + threadIdx.x;
    int e = i >> 5;
    if (e >= NE) return;
    int c0 = (i & 31) * 4;
    int h = c0 >> shift;
    int d = dst[e], s = src[e];
    int idx = d * H + h;
    float alpha = __expf(sbuf[(size_t)e * H + h] - dec_f(smaxi[idx])) / den[idx];
    const float4 v = *(const float4*)(xl + (size_t)s * DIM + c0);
    float* o = hbuf + (size_t)d * DIM + c0;
    atomicAdd(o + 0, v.x * alpha);
    atomicAdd(o + 1, v.y * alpha);
    atomicAdd(o + 2, v.z * alpha);
    atomicAdd(o + 3, v.w * alpha);
}

// ---- add bias, LayerNorm, ReLU; emit bf16 for next layer (and f32 output) ----
__global__ void __launch_bounds__(256) k_ln_relu(
    const float* __restrict__ hbuf, const float* __restrict__ b,
    const float* __restrict__ g, const float* __restrict__ be,
    __bf16* __restrict__ xnext, float* __restrict__ outf, int write_out) {
    int wave = threadIdx.x >> 5, lane = threadIdx.x & 31;
    int node = blockIdx.x * 8 + wave;
    if (node >= NN) return;
    int c0 = lane * 4;
    float4 v = *(const float4*)(hbuf + (size_t)node * DIM + c0);
    v.x += b[c0]; v.y += b[c0 + 1]; v.z += b[c0 + 2]; v.w += b[c0 + 3];
    float s  = v.x + v.y + v.z + v.w;
    float sq = v.x * v.x + v.y * v.y + v.z * v.z + v.w * v.w;
    for (int m = 16; m >= 1; m >>= 1) { s += __shfl_xor(s, m); sq += __shfl_xor(sq, m); }
    float mean = s * (1.f / 128.f);
    float var  = sq * (1.f / 128.f) - mean * mean;
    float inv  = rsqrtf(var + 1e-5f);
    float y[4] = {v.x, v.y, v.z, v.w};
    float4 ov;
    float* op = &ov.x;
#pragma unroll
    for (int j = 0; j < 4; ++j) {
        float t = (y[j] - mean) * inv * g[c0 + j] + be[c0 + j];
        t = t > 0.f ? t : 0.f;
        op[j] = t;
        xnext[(size_t)node * DIM + c0 + j] = (__bf16)t;
    }
    if (write_out) *(float4*)(outf + (size_t)node * DIM + c0) = ov;
}

extern "C" void kernel_launch(void* const* d_in, const int* in_sizes, int n_in,
                              void* d_out, int out_size, void* d_ws, size_t ws_size,
                              hipStream_t stream) {
    const float* x   = (const float*)d_in[0];
    const float* ea  = (const float*)d_in[1];
    const int*   src = (const int*)d_in[2];
    const int*   dst = (const int*)d_in[3];

    char* ws = (char*)d_ws;
    size_t off = 0;
    auto alloc = [&](size_t bytes) -> void* {
        off = (off + 255) & ~(size_t)255;
        void* p = ws + off; off += bytes; return p;
    };
    __bf16* xbf   = (__bf16*)alloc((size_t)NN * DIM * 2);
    __bf16* eabf  = (__bf16*)alloc((size_t)NE * EDIM * 2);
    __bf16* Wlt   = (__bf16*)alloc((size_t)DIM * DIM * 2);
    __bf16* Wrt   = (__bf16*)alloc((size_t)DIM * DIM * 2);
    __bf16* Wet   = (__bf16*)alloc((size_t)DIM * EDIM * 2);
    float*  xl    = (float*)alloc((size_t)NN * DIM * 4);
    float*  xr    = (float*)alloc((size_t)NN * DIM * 4);
    float*  sbuf  = (float*)alloc((size_t)NE * 4 * 4);
    int*    smaxi = (int*)  alloc((size_t)NN * 4 * 4);
    float*  den   = (float*)alloc((size_t)NN * 4 * 4);
    float*  hbuf  = (float*)alloc((size_t)NN * DIM * 4);

    // one-time casts
    k_cast_bf16<<<(NN * DIM + 255) / 256, 256, 0, stream>>>(x, xbf, NN * DIM);
    k_cast_bf16<<<(NE * EDIM + 255) / 256, 256, 0, stream>>>(ea, eabf, NE * EDIM);

    for (int l = 0; l < 3; ++l) {
        int base = 4 + 9 * l;
        const float* Wl  = (const float*)d_in[base + 0];
        const float* bl  = (const float*)d_in[base + 1];
        const float* Wr  = (const float*)d_in[base + 2];
        const float* br  = (const float*)d_in[base + 3];
        const float* We  = (const float*)d_in[base + 4];
        const float* att = (const float*)d_in[base + 5];
        const float* bb  = (const float*)d_in[base + 6];
        const float* gg  = (const float*)d_in[base + 7];
        const float* be  = (const float*)d_in[base + 8];
        int H = (l == 2) ? 1 : 4;
        int shift = (l == 2) ? 7 : 5;

        // transpose+cast weights
        k_transpose_cast<<<(DIM * DIM + 255) / 256, 256, 0, stream>>>(Wl, Wlt, DIM, DIM);
        k_transpose_cast<<<(DIM * DIM + 255) / 256, 256, 0, stream>>>(Wr, Wrt, DIM, DIM);
        k_transpose_cast<<<(EDIM * DIM + 255) / 256, 256, 0, stream>>>(We, Wet, EDIM, DIM);

        // node transforms via WMMA
        int gemm_grid = (NN / 16 + 7) / 8;
        k_gemm_node<<<gemm_grid, 256, 0, stream>>>(xbf, Wlt, bl, xl, NN);
        k_gemm_node<<<gemm_grid, 256, 0, stream>>>(xbf, Wrt, br, xr, NN);

        // init accumulators
        k_init<<<(NN * DIM + 255) / 256, 256, 0, stream>>>(smaxi, den, hbuf, NN * H);

        // fused edge projection (WMMA) + attention logits
        k_edge_score<<<NE / 64, 128, 0, stream>>>(eabf, Wet, xl, xr, src, dst,
                                                  att, sbuf, H, shift);

        // segment softmax
        int eh = NE * H;
        k_seg_max<<<(eh + 255) / 256, 256, 0, stream>>>(sbuf, dst, smaxi, H);
        k_seg_den<<<(eh + 255) / 256, 256, 0, stream>>>(sbuf, dst, smaxi, den, H);

        // weighted aggregation
        k_aggregate<<<(NE * 32) / 256, 256, 0, stream>>>(sbuf, src, dst, smaxi, den,
                                                         xl, hbuf, H, shift);

        // +bias, LN, ReLU -> next-layer bf16 input (and final f32 output)
        k_ln_relu<<<(NN + 7) / 8, 256, 0, stream>>>(hbuf, bb, gg, be, xbf,
                                                    (float*)d_out, l == 2 ? 1 : 0);
    }
}
